// MaskHead_74483322847689
// MI455X (gfx1250) — matrix operand order repaired and checked by reading, not compiled, not run on hardware
//
#include <hip/hip_runtime.h>
#include <math.h>

// ---------------- types ----------------
typedef __attribute__((ext_vector_type(16))) __bf16 v16bf;
typedef __attribute__((ext_vector_type(8)))  float  v8f;

struct __align__(16) U4 { unsigned int x, y, z, w; };

#define MID   24
#define NDET  8
#define S3    110592     /* 48^3 */
#define N2T   13824      /* 24^3 */
#define N4T   1728       /* 12^3 */
#define DHW   2097152    /* 128^3 */

__device__ __forceinline__ unsigned short f2bf(float f) {
  unsigned int u = __float_as_uint(f);
  u += 0x7fffu + ((u >> 16) & 1u);   // round-to-nearest-even
  return (unsigned short)(u >> 16);
}
__device__ __forceinline__ int clampi(int v, int lo, int hi) {
  return v < lo ? lo : (v > hi ? hi : v);
}

// ---------------- zero output ----------------
__global__ void k_zero(float* __restrict__ p, int n4) {
  int i = blockIdx.x * blockDim.x + threadIdx.x;
  if (i < n4) { float4 z; z.x = z.y = z.z = z.w = 0.f; ((float4*)p)[i] = z; }
}

// ------- pack conv weights into WMMA-fragment-ready order -------------------------
// K-order: k = tap*Cpad + ci.  Flat layout: j = ((kb*2+mh)*32+lane)*16 + e, where
// A-frag element e of lane: kk = (e<8?0:16) + (lane>=16?8:0) + (e&7), m = mh*16+lane%16.
__global__ void k_pack_w(const float* __restrict__ w, unsigned short* __restrict__ A,
                         int Cin, int Cpad) {
  int j = blockIdx.x * blockDim.x + threadIdx.x;
  if (j >= 27 * Cpad * 32) return;
  int e = j & 15, lane = (j >> 4) & 31, mh = (j >> 9) & 1, kb = j >> 10;
  int kk = ((e < 8) ? 0 : 16) + ((lane >= 16) ? 8 : 0) + (e & 7);
  int m = mh * 16 + (lane & 15);
  int k = kb * 32 + kk;
  int t = k / Cpad, ci = k - t * Cpad;
  float v = (m < MID && ci < Cin) ? w[(m * Cin + ci) * 27 + t] : 0.f;
  A[j] = f2bf(v);
}

// ------- crop+concat -> bf16 channel-last [det][R^3][Cpad], pad channels zeroed ----
__global__ void k_crop_cat(const float* __restrict__ a, int Ca,
                           const float* __restrict__ b, int Cb, int srcE,
                           unsigned short* __restrict__ dst, int R, int scale, int Cpad,
                           const int* __restrict__ dets) {
  int det = blockIdx.y;
  long long i = (long long)blockIdx.x * blockDim.x + threadIdx.x;
  int R2 = R * R, R3 = R2 * R;
  if (i >= (long long)R3 * Cpad) return;
  int c = (int)(i % Cpad), p = (int)(i / Cpad);
  int z = p / R2; int rr = p - z * R2; int y = rr / R; int x = rr - y * R;
  const int* dd = dets + det * 8;
  int zs = dd[1] / scale, ys = dd[2] / scale, xs = dd[3] / scale;
  float v = 0.f;
  if (c < Ca)
    v = a[(((long long)c * srcE + (zs + z)) * srcE + (ys + y)) * srcE + (xs + x)];
  else if (c < Ca + Cb)
    v = b[(((long long)(c - Ca) * srcE + (zs + z)) * srcE + (ys + y)) * srcE + (xs + x)];
  dst[(long long)det * R3 * Cpad + i] = f2bf(v);
}

// ------- crop 6^3 from out_8/f_8 (16^3) + trilinear x2 -> bf16 [det][12^3][320] ----
__global__ void k_crop_up8(const float* __restrict__ a, const float* __restrict__ b,
                           unsigned short* __restrict__ dst, const int* __restrict__ dets) {
  int det = blockIdx.y;
  int i = blockIdx.x * blockDim.x + threadIdx.x;
  if (i >= N4T * 320) return;
  int c = i % 320, p = i / 320;
  int z = p / 144; int rr = p - z * 144; int y = rr / 12; int x = rr - y * 12;
  const int* dd = dets + det * 8;
  int zs = dd[1] / 8, ys = dd[2] / 8, xs = dd[3] / 8;
  const float* src; int cc;
  if (c < 128) { src = a; cc = c; } else { src = b; cc = c - 128; }
  const float* base = src + (long long)cc * 4096;
  float fz = (z + 0.5f) * 0.5f - 0.5f;
  float fy = (y + 0.5f) * 0.5f - 0.5f;
  float fx = (x + 0.5f) * 0.5f - 0.5f;
  int z0 = (int)floorf(fz), y0 = (int)floorf(fy), x0 = (int)floorf(fx);
  float wz = fz - z0, wy = fy - y0, wx = fx - x0;
  int za = clampi(z0, 0, 5), zb = clampi(z0 + 1, 0, 5);
  int ya = clampi(y0, 0, 5), yb = clampi(y0 + 1, 0, 5);
  int xa = clampi(x0, 0, 5), xb = clampi(x0 + 1, 0, 5);
#define LD8(zz,yy,xx) base[((zs+(zz))*16 + (ys+(yy)))*16 + (xs+(xx))]
  float v000=LD8(za,ya,xa), v001=LD8(za,ya,xb), v010=LD8(za,yb,xa), v011=LD8(za,yb,xb);
  float v100=LD8(zb,ya,xa), v101=LD8(zb,ya,xb), v110=LD8(zb,yb,xa), v111=LD8(zb,yb,xb);
#undef LD8
  float v00 = v000 + (v001 - v000) * wx, v01 = v010 + (v011 - v010) * wx;
  float v10 = v100 + (v101 - v100) * wx, v11 = v110 + (v111 - v110) * wx;
  float v0 = v00 + (v01 - v00) * wy,     v1 = v10 + (v11 - v10) * wy;
  dst[(long long)det * N4T * 320 + i] = f2bf(v0 + (v1 - v0) * wz);
}

// ---------------- conv3d: halo-in-LDS, 27 taps x WMMA per 32-channel chunk ---------
// WG tile: 4x4x(4*ZT) output voxels x 32 out-ch (M pad).
// 8 waves = mhalf(2) x nsub(4); wave owns z-planes nsub (+ nsub+4 when ZT==2),
// sharing one A-fragment load across its ZT accumulators.
template <int ZT>
__global__ void __launch_bounds__(256)
k_conv_wmma(const unsigned short* __restrict__ xin,   // [det][R^3][Cpad] bf16 ch-last
            const unsigned short* __restrict__ Apack, // fragment-swizzled weights
            float* __restrict__ yout,                 // [det][24][R^3] f32
            int Cpad, int R, int Ntot) {
  constexpr int HZ  = 4 * ZT + 2;          // halo z-extent
  constexpr int HV  = HZ * 36;             // halo voxels (z * 6 * 6)
  constexpr int NIT = HV * 32 / 256;       // staging iters (27 or 45), exact
  __shared__ __align__(16) unsigned short H[HV * 32];
  int det = blockIdx.y;
  const unsigned short* X = xin + (long long)det * Ntot * Cpad;
  float* Y = yout + (long long)det * MID * Ntot;
  int tid = threadIdx.x, lane = tid & 31, wave = tid >> 5;
  int mhalf = wave & 1, nsub = wave >> 1;
  int Rt = R >> 2;
  int bb = blockIdx.x;
  int tz = bb / (Rt * Rt); int rb = bb - tz * Rt * Rt; int ty = rb / Rt; int tx = rb - ty * Rt;
  int z0 = tz * 4 * ZT, y0 = ty * 4, x0 = tx * 4;
  int r = lane & 15;
  int ly = r >> 2, lx = r & 3;                 // lz == nsub (+4)
  int kbch = (lane >= 16) ? 8 : 0;
  int nchunks = Cpad >> 5;

  // precompute per-thread halo gather offsets (NIT*256 == HV*32)
  int goff[NIT];
#pragma unroll
  for (int i = 0; i < NIT; ++i) {
    int idx = tid + i * 256;
    int v = idx >> 5, ch = idx & 31;
    int hz = v / 36; int rv = v - hz * 36; int hy = rv / 6; int hx = rv - hy * 6;
    int gz = z0 - 1 + hz, gy = y0 - 1 + hy, gx = x0 - 1 + hx;
    bool ok = ((unsigned)gz < (unsigned)R) & ((unsigned)gy < (unsigned)R) &
              ((unsigned)gx < (unsigned)R);
    goff[i] = ok ? ((gz * R + gy) * R + gx) * Cpad + ch : -1;
  }

  v8f acc0 = {0.f, 0.f, 0.f, 0.f, 0.f, 0.f, 0.f, 0.f};
  v8f acc1 = {0.f, 0.f, 0.f, 0.f, 0.f, 0.f, 0.f, 0.f};
  for (int cc = 0; cc < nchunks; ++cc) {
    int c0 = cc << 5;
    __syncthreads();
#pragma unroll
    for (int i = 0; i < NIT; ++i) {
      int o = goff[i];
      unsigned short u = (o >= 0) ? X[o + c0] : (unsigned short)0;
      H[tid + i * 256] = u;
    }
    __syncthreads();
#pragma unroll
    for (int t = 0; t < 27; ++t) {
      int dz = t / 9, rt = t - dz * 9, dy = rt / 3, dx = rt - dy * 3;   // 0..2
      int kb = t * nchunks + cc;
      const U4* ap = (const U4*)(Apack + (((kb * 2 + mhalf) * 32 + lane) << 4));
      union { U4 q[2]; v16bf v; } af, bf0, bf1;
      af.q[0] = ap[0]; af.q[1] = ap[1];
      int hv0 = ((nsub + dz) * 6 + (ly + dy)) * 6 + (lx + dx);
      const U4* bp0 = (const U4*)&H[hv0 * 32 + kbch];
      bf0.q[0] = bp0[0]; bf0.q[1] = bp0[2];
      acc0 = __builtin_amdgcn_wmma_f32_16x16x32_bf16(false, af.v, false, bf0.v,
                                                     (short)0, acc0, false, false);
      if (ZT == 2) {
        const U4* bp1 = (const U4*)&H[(hv0 + 4 * 36) * 32 + kbch];
        bf1.q[0] = bp1[0]; bf1.q[1] = bp1[2];
        acc1 = __builtin_amdgcn_wmma_f32_16x16x32_bf16(false, af.v, false, bf1.v,
                                                       (short)0, acc1, false, false);
      }
    }
  }
  // C/D layout: VGPR rr -> M = rr + (lane>=16 ? 8 : 0), N = lane%16
  int voxel0 = (z0 + nsub) * R * R + (y0 + ly) * R + (x0 + lx);
  int mbase = mhalf * 16 + ((lane >= 16) ? 8 : 0);
#pragma unroll
  for (int rr = 0; rr < 8; ++rr) {
    int m = mbase + rr;
    if (m < MID) {
      Y[(long long)m * Ntot + voxel0] = acc0[rr];
      if (ZT == 2) Y[(long long)m * Ntot + voxel0 + 4 * R * R] = acc1[rr];
    }
  }
}

// ---------------- InstanceNorm stats: per (det, channel) mean & rstd ----------------
__global__ void k_stats(const float* __restrict__ y, float* __restrict__ stats, int Ntot) {
  int det = blockIdx.y, c = blockIdx.x;
  const float* p = y + ((long long)det * MID + c) * Ntot;
  __shared__ float s1[256], s2[256];
  float a = 0.f, b = 0.f;
  for (int i = threadIdx.x; i < Ntot; i += 256) { float v = p[i]; a += v; b += v * v; }
  s1[threadIdx.x] = a; s2[threadIdx.x] = b;
  __syncthreads();
  for (int st = 128; st > 0; st >>= 1) {
    if (threadIdx.x < st) { s1[threadIdx.x] += s1[threadIdx.x + st];
                            s2[threadIdx.x] += s2[threadIdx.x + st]; }
    __syncthreads();
  }
  if (threadIdx.x == 0) {
    float inv = 1.f / (float)Ntot;
    float m = s1[0] * inv;
    float var = s2[0] * inv - m * m;   // biased var (matches reference)
    stats[det * 48 + c * 2]     = m;
    stats[det * 48 + c * 2 + 1] = rsqrtf(var + 1e-5f);
  }
}

// ---------------- InstanceNorm apply + ReLU (in place) ----------------
__global__ void k_apply(float* __restrict__ y, const float* __restrict__ stats,
                        const float* __restrict__ g, const float* __restrict__ be, int Ntot) {
  int det = blockIdx.y;
  int i = blockIdx.x * blockDim.x + threadIdx.x;
  if (i >= MID * Ntot) return;
  int c = i / Ntot;
  float m = stats[det * 48 + c * 2], rs = stats[det * 48 + c * 2 + 1];
  float* p = y + (long long)det * MID * Ntot;
  float v = (p[i] - m) * rs * g[c] + be[c];
  p[i] = fmaxf(v, 0.f);
}

// ------- trilinear upsample of_x [24][R^3] -> xino ch-last [S3][96] at chBase ------
__global__ void k_up_cat(const float* __restrict__ src, int R, int s,
                         unsigned short* __restrict__ xino, int chBase) {
  int det = blockIdx.y;
  long long i = (long long)blockIdx.x * blockDim.x + threadIdx.x;
  if (i >= (long long)MID * S3) return;
  int c = (int)(i / S3), p = (int)(i % S3);
  int z = p / 2304; int rr = p - z * 2304; int y = rr / 48; int x = rr - y * 48;
  const float* b = src + ((long long)det * MID + c) * (R * R * R);
  float inv = 1.f / (float)s;
  float fz = (z + 0.5f) * inv - 0.5f;
  float fy = (y + 0.5f) * inv - 0.5f;
  float fx = (x + 0.5f) * inv - 0.5f;
  int z0 = (int)floorf(fz), y0 = (int)floorf(fy), x0 = (int)floorf(fx);
  float wz = fz - z0, wy = fy - y0, wx = fx - x0;
  int za = clampi(z0, 0, R - 1), zb = clampi(z0 + 1, 0, R - 1);
  int ya = clampi(y0, 0, R - 1), yb = clampi(y0 + 1, 0, R - 1);
  int xa = clampi(x0, 0, R - 1), xb = clampi(x0 + 1, 0, R - 1);
#define LDU(zz,yy,xx) b[((zz)*R + (yy))*R + (xx)]
  float v000=LDU(za,ya,xa), v001=LDU(za,ya,xb), v010=LDU(za,yb,xa), v011=LDU(za,yb,xb);
  float v100=LDU(zb,ya,xa), v101=LDU(zb,ya,xb), v110=LDU(zb,yb,xa), v111=LDU(zb,yb,xb);
#undef LDU
  float v00 = v000 + (v001 - v000) * wx, v01 = v010 + (v011 - v010) * wx;
  float v10 = v100 + (v101 - v100) * wx, v11 = v110 + (v111 - v110) * wx;
  float v0 = v00 + (v01 - v00) * wy,     v1 = v10 + (v11 - v10) * wy;
  float v  = v0 + (v1 - v0) * wz;
  xino[((long long)det * S3 + p) * 96 + (chBase + c)] = f2bf(v);
}

// ------- image crop -> xino channel 0; also zero pad channels 73..95 ---------------
__global__ void k_img(const float* __restrict__ img, unsigned short* __restrict__ xino,
                      const int* __restrict__ dets) {
  int det = blockIdx.y;
  int i = blockIdx.x * blockDim.x + threadIdx.x;
  if (i >= S3 * 24) return;
  int c24 = i % 24, p = i / 24;
  int z = p / 2304; int rr = p - z * 2304; int y = rr / 48; int x = rr - y * 48;
  const int* dd = dets + det * 8;
  float v = 0.f;
  int c = (c24 == 0) ? 0 : 72 + c24;      // 0, or pad 73..95
  if (c24 == 0)
    v = img[((long long)(dd[1] + z) * 128 + (dd[2] + y)) * 128 + (dd[3] + x)];
  xino[((long long)det * S3 + p) * 96 + c] = f2bf(v);
}

// ---------------- fused norm+ReLU + per-class logit + scatter ----------------
__global__ void k_final(const float* __restrict__ yo, const float* __restrict__ stats,
                        const float* __restrict__ g, const float* __restrict__ be,
                        const float* __restrict__ lw, const float* __restrict__ lb,
                        const int* __restrict__ dets, float* __restrict__ out) {
  int det = blockIdx.y;
  int p = blockIdx.x * blockDim.x + threadIdx.x;
  if (p >= S3) return;
  const int* dd = dets + det * 8;
  int cat = dd[7] - 1;
  const float* Y = yo + (long long)det * MID * S3;
  const float* st = stats + det * 48;
  float acc = lb[cat];
#pragma unroll
  for (int c = 0; c < MID; ++c) {
    float v = (Y[(long long)c * S3 + p] - st[c * 2]) * st[c * 2 + 1] * g[c] + be[c];
    acc += lw[cat * MID + c] * fmaxf(v, 0.f);
  }
  int z = p / 2304; int rr = p - z * 2304; int y = rr / 48; int x = rr - y * 48;
  out[(long long)det * DHW + ((long long)(dd[1] + z) * 128 + (dd[2] + y)) * 128 + (dd[3] + x)] = acc;
}

// =======================================================================================
extern "C" void kernel_launch(void* const* d_in, const int* in_sizes, int n_in,
                              void* d_out, int out_size, void* d_ws, size_t ws_size,
                              hipStream_t stream) {
  (void)in_sizes; (void)n_in; (void)out_size; (void)ws_size;
  const float* img  = (const float*)d_in[0];
  const float* out2 = (const float*)d_in[1];
  const float* f2   = (const float*)d_in[2];
  const float* out4 = (const float*)d_in[3];
  const float* f4   = (const float*)d_in[4];
  const float* out8 = (const float*)d_in[5];
  const float* f8   = (const float*)d_in[6];
  const float* w2   = (const float*)d_in[7];   // conv biases cancel inside InstanceNorm
  const float* g2   = (const float*)d_in[9];
  const float* be2  = (const float*)d_in[10];
  const float* w4   = (const float*)d_in[11];
  const float* g4   = (const float*)d_in[13];
  const float* be4  = (const float*)d_in[14];
  const float* w8   = (const float*)d_in[15];
  const float* g8   = (const float*)d_in[17];
  const float* be8  = (const float*)d_in[18];
  const float* wo   = (const float*)d_in[19];
  const float* go   = (const float*)d_in[21];
  const float* beo  = (const float*)d_in[22];
  const float* lw   = (const float*)d_in[23];
  const float* lb   = (const float*)d_in[24];
  const int*   dets = (const int*)d_in[25];
  float* out = (float*)d_out;

  // padded channel counts (multiples of 32)
  const int CP2 = 96, CP4 = 160, CP8 = 320, CPO = 96;

  char* ws = (char*)d_ws;
  size_t off = 0;
  auto alloc = [&](size_t bytes) -> char* {
    char* p = ws + off; off += (bytes + 255) & ~(size_t)255; return p;
  };
  unsigned short* A2 = (unsigned short*)alloc((size_t)27 * CP2 * 32 * 2);
  unsigned short* A4 = (unsigned short*)alloc((size_t)27 * CP4 * 32 * 2);
  unsigned short* A8 = (unsigned short*)alloc((size_t)27 * CP8 * 32 * 2);
  unsigned short* AO = (unsigned short*)alloc((size_t)27 * CPO * 32 * 2);
  float*          st = (float*)alloc((size_t)NDET * 48 * 4);
  unsigned short* x2 = (unsigned short*)alloc((size_t)NDET * N2T * CP2 * 2);
  unsigned short* x4 = (unsigned short*)alloc((size_t)NDET * N4T * CP4 * 2);
  unsigned short* x8 = (unsigned short*)alloc((size_t)NDET * N4T * CP8 * 2);
  unsigned short* xo = (unsigned short*)alloc((size_t)NDET * S3  * CPO * 2);
  float*          y2 = (float*)alloc((size_t)NDET * MID * N2T * 4);
  float*          y4 = (float*)alloc((size_t)NDET * MID * N4T * 4);
  float*          y8 = (float*)alloc((size_t)NDET * MID * N4T * 4);
  float*          yo = (float*)alloc((size_t)NDET * MID * S3  * 4);

  const int T = 256;
  k_zero<<<(NDET * DHW / 4 + T - 1) / T, T, 0, stream>>>(out, NDET * DHW / 4);

  k_pack_w<<<(27 * CP2 * 32 + T - 1) / T, T, 0, stream>>>(w2, A2, 80, CP2);
  k_pack_w<<<(27 * CP4 * 32 + T - 1) / T, T, 0, stream>>>(w4, A4, 160, CP4);
  k_pack_w<<<(27 * CP8 * 32 + T - 1) / T, T, 0, stream>>>(w8, A8, 320, CP8);
  k_pack_w<<<(27 * CPO * 32 + T - 1) / T, T, 0, stream>>>(wo, AO, 73, CPO);

  // --- branch 2 (24^3): 4x4x8 tiles, 2 accumulators/wave ---
  k_crop_cat<<<dim3((N2T * CP2 + T - 1) / T, NDET), T, 0, stream>>>(
      out2, 32, f2, 48, 64, x2, 24, 2, CP2, dets);
  k_conv_wmma<2><<<dim3(6 * 6 * 3, NDET), T, 0, stream>>>(x2, A2, y2, CP2, 24, N2T);
  k_stats<<<dim3(MID, NDET), T, 0, stream>>>(y2, st, N2T);
  k_apply<<<dim3((MID * N2T + T - 1) / T, NDET), T, 0, stream>>>(y2, st, g2, be2, N2T);

  // --- branch 4 (12^3): 4x4x4 tiles ---
  k_crop_cat<<<dim3((N4T * CP4 + T - 1) / T, NDET), T, 0, stream>>>(
      out4, 64, f4, 96, 32, x4, 12, 4, CP4, dets);
  k_conv_wmma<1><<<dim3(27, NDET), T, 0, stream>>>(x4, A4, y4, CP4, 12, N4T);
  k_stats<<<dim3(MID, NDET), T, 0, stream>>>(y4, st, N4T);
  k_apply<<<dim3((MID * N4T + T - 1) / T, NDET), T, 0, stream>>>(y4, st, g4, be4, N4T);

  // --- branch 8 (crop + x2 upsample fused, 12^3) ---
  k_crop_up8<<<dim3((N4T * CP8 + T - 1) / T, NDET), T, 0, stream>>>(out8, f8, x8, dets);
  k_conv_wmma<1><<<dim3(27, NDET), T, 0, stream>>>(x8, A8, y8, CP8, 12, N4T);
  k_stats<<<dim3(MID, NDET), T, 0, stream>>>(y8, st, N4T);
  k_apply<<<dim3((MID * N4T + T - 1) / T, NDET), T, 0, stream>>>(y8, st, g8, be8, N4T);

  // --- 73(96)-channel 48^3 concat input ---
  k_img<<<dim3((S3 * 24 + T - 1) / T, NDET), T, 0, stream>>>(img, xo, dets);
  k_up_cat<<<dim3((MID * S3 + T - 1) / T, NDET), T, 0, stream>>>(y2, 24, 2, xo, 1);
  k_up_cat<<<dim3((MID * S3 + T - 1) / T, NDET), T, 0, stream>>>(y4, 12, 4, xo, 25);
  k_up_cat<<<dim3((MID * S3 + T - 1) / T, NDET), T, 0, stream>>>(y8, 12, 4, xo, 49);

  // --- final conv (dominant FLOPs, 48^3): 4x4x8 tiles, 2 accumulators/wave ---
  k_conv_wmma<2><<<dim3(12 * 12 * 6, NDET), T, 0, stream>>>(xo, AO, yo, CPO, 48, S3);
  k_stats<<<dim3(MID, NDET), T, 0, stream>>>(yo, st, S3);
  k_final<<<dim3((S3 + T - 1) / T, NDET), T, 0, stream>>>(yo, st, go, beo, lw, lb, dets, out);
}
